// LocallyConnected2d_binary_6983616823395
// MI455X (gfx1250) — matrix operand order repaired and checked by reading, not compiled, not run on hardware
//
#include <hip/hip_runtime.h>

typedef __attribute__((ext_vector_type(16))) __bf16 v16bf;
typedef __attribute__((ext_vector_type(8)))  __bf16 v8bf;
typedef __attribute__((ext_vector_type(8)))  float  v8f;
typedef __attribute__((ext_vector_type(4)))  unsigned int u32x4;
typedef __attribute__((ext_vector_type(8)))  int    i32x8;
typedef __attribute__((ext_vector_type(4)))  int    i32x4;

#define BB   32
#define CIN  64
#define COUT 64
#define HH   32
#define WW   32
#define K2   9
#define NKC  32                          // 32 chunks of K=32 (2 channels, 9->16 pad)
#define NTOT (BB * COUT * HH * WW)       // 2097152
#define NWEL ((long)COUT * CIN * HH * WW * K2) // 37748736
#define BN_EPS 1e-5f

// dynamic LDS layout (bytes) for the conv kernel
#define LDS_BW0   0                      // Bw buf0: 64*2*9 f32 = 4608 B
#define LDS_BW1   4608                   // Bw buf1
#define LDS_AS    9216                   // As: 32 rows * 40 bf16 = 2560 B
#define LDS_TOTAL (9216 + 2560)
#define BW_ELEMS  1152                   // 64*18 f32 per buffer

// workspace layout (floats)
#define WS_SCALEW   0
#define WS_SCALEOUT 1
#define WS_P1       16
#define WS_P5       (16 + 2048)
#define WS_CHA      (16 + 4096)
#define WS_CHB      (16 + 4096 + 64)
#define NB_RED      2048

#if __has_builtin(__builtin_amdgcn_tensor_load_to_lds) && \
    __has_builtin(__builtin_amdgcn_s_wait_tensorcnt)
#define HAVE_TDM 1
#else
#define HAVE_TDM 0
#endif

// ---------------------------------------------------------------- reductions
__global__ __launch_bounds__(256) void k_absw(const float* __restrict__ w,
                                              float* __restrict__ ws) {
  __shared__ float red[256];
  float s = 0.f;
  for (long i = (long)blockIdx.x * 256 + threadIdx.x; i < NWEL;
       i += (long)NB_RED * 256)
    s += fabsf(w[i]);
  red[threadIdx.x] = s;
  __syncthreads();
  for (int st = 128; st > 0; st >>= 1) {
    if ((int)threadIdx.x < st) red[threadIdx.x] += red[threadIdx.x + st];
    __syncthreads();
  }
  if (threadIdx.x == 0) ws[WS_P1 + blockIdx.x] = red[0];
}

__global__ __launch_bounds__(256) void k_finalize(const float* __restrict__ part,
                                                  float* __restrict__ dst,
                                                  float inv) {
  __shared__ float red[256];
  float s = 0.f;
  for (int i = threadIdx.x; i < NB_RED; i += 256) s += part[i];
  red[threadIdx.x] = s;
  __syncthreads();
  for (int st = 128; st > 0; st >>= 1) {
    if ((int)threadIdx.x < st) red[threadIdx.x] += red[threadIdx.x + st];
    __syncthreads();
  }
  if (threadIdx.x == 0) *dst = red[0] * inv;
}

// ---------------------------------------------------------------- main WMMA
// One workgroup per (h,w); 8 waves, one 16x16 output tile each.
// K padded to 16/channel -> 32 chunks of K=32 covering channels (2kc, 2kc+1).
// Weights for each chunk arrive by TDM tensor_load_to_lds (double-buffered):
//   3-D tile: dim0 = 9 f32 contiguous (kk), dim1 = 2 (c, stride 9216 elem),
//   dim2 = 64 (o, stride 589824 elem) -> packed [o][c_local][kk] f32 in LDS.
__global__ __launch_bounds__(256) void k_conv_wmma(
    const float* __restrict__ x, const float* __restrict__ w,
    const float* __restrict__ bias, const float* __restrict__ ws,
    float* __restrict__ out) {
  extern __shared__ char smem[];
  float*  Bw = (float*)smem;                   // [2][1152] f32
  __bf16* As = (__bf16*)(smem + LDS_AS);       // [32][40] bf16 (80B rows)

  const int hw = blockIdx.x;
  const int h = hw >> 5, wq = hw & 31;
  const int tid = threadIdx.x;
  const int lane = tid & 31, wid = tid >> 5;
  const int mt = wid >> 2, nt = wid & 3;
  const int hi = (lane < 16) ? 0 : 1;          // half-wave selector
  const int arow = mt * 16 + (lane & 15);      // batch row of A fragment
  const int brow = nt * 16 + (lane & 15);      // cout column of B fragment
  const float scaleW = ws[WS_SCALEW];

  v8f acc = {};

#if HAVE_TDM
  auto tdm_issue = [&](int kc, int buf) {
    unsigned long long ga = (unsigned long long)(const void*)w +
        4ull * (unsigned long long)(((2 * kc * 32 + h) * 32 + wq) * 9);
    u32x4 g0 = {1u,                                   // count=1 (user D#)
                (unsigned)(buf ? LDS_BW1 : LDS_BW0),  // lds_addr
                (unsigned)(ga & 0xffffffffull),       // global_addr[31:0]
                (unsigned)((ga >> 32) & 0x1ffffffull) | (2u << 30)}; // + type=2
    // g1: data_size=2(4B); tensor_dim0=9, tensor_dim1=2;
    //     tile_dim0=9, tile_dim1=2, tile_dim2=64;
    //     tensor_dim0_stride=9216, tensor_dim1_stride=589824 (=9<<16 in w7)
    i32x8 g1 = {0x00020000, 0x00090000, 0x00020000, 0x00090000,
                (64 << 16) | 2, 9216, 0, 9};
    i32x4 g2 = {64, 0, 0, 0};                         // tensor_dim2=64
    i32x4 g3 = {0, 0, 0, 0};
#if __clang_major__ >= 23
    i32x8 g4 = {0, 0, 0, 0, 0, 0, 0, 0};
    __builtin_amdgcn_tensor_load_to_lds(g0, g1, g2, g3, g4, 0);
#else
    __builtin_amdgcn_tensor_load_to_lds(g0, g1, g2, g3, 0);
#endif
  };
  if (wid == 0) tdm_issue(0, 0);                      // prologue: chunk 0
#endif

  for (int kc = 0; kc < NKC; ++kc) {
    const int buf = kc & 1;
    // ---- stage A patches: 32 batch rows x 32 padded-K, bf16
#pragma unroll
    for (int i = 0; i < 4; ++i) {
      int e = tid * 4 + i;                    // 0..1023
      int b = e >> 5, j = e & 31;
      int cl = j >> 4, kk = j & 15;           // channel-local, tap (pad >= 9)
      int c = 2 * kc + cl;
      float v = 0.f;
      if (kk < 9) {
        int h2 = h + kk / 3 - 1;
        int w2 = wq + (kk % 3) - 1;
        if ((unsigned)h2 < 32u && (unsigned)w2 < 32u)
          v = x[((b * CIN + c) * HH + h2) * WW + w2];
        if (i == 0 && kc + 1 < NKC)           // keep prefetch path alive
          __builtin_prefetch(&x[((b * CIN + c + 2) * HH + h) * WW + wq], 0, 0);
      }
      As[b * 40 + j] = (__bf16)v;
    }

#if HAVE_TDM
    if (wid == 0) {
      if (kc + 1 < NKC) {
        tdm_issue(kc + 1, buf ^ 1);           // pipeline next chunk
        __builtin_amdgcn_s_wait_tensorcnt(1); // wait: current chunk landed
      } else {
        __builtin_amdgcn_s_wait_tensorcnt(0);
      }
    }
#else
    // fallback: plain global staging of this chunk's weights
    for (int i = tid; i < BW_ELEMS; i += 256) {
      int o = i / 18, r = i - o * 18, cl = r / 9, kk = r - cl * 9;
      long idx = (((long)(o * CIN + (2 * kc + cl)) * HH + h) * WW + wq) * K2 + kk;
      Bw[buf * BW_ELEMS + i] = w[idx];
    }
#endif
    __syncthreads();

    // ---- A fragment: lanes 0-15 hold K 0-15; lanes 16-31: K 16-31
    const __bf16* ap = &As[arow * 40 + hi * 16];
    v8bf alo = *(const v8bf*)ap;
    v8bf ahi = *(const v8bf*)(ap + 8);
    v16bf a = __builtin_shufflevector(alo, ahi, 0, 1, 2, 3, 4, 5, 6, 7, 8, 9,
                                      10, 11, 12, 13, 14, 15);
    // ---- B fragment: lane n = column o; K half = channel-local hi;
    //      9 consecutive f32 -> sign bf16, pad slots 9..15 = 0
    const float* bp = &Bw[buf * BW_ELEMS + brow * 18 + hi * 9];
    v16bf bfv = {};
#pragma unroll
    for (int k = 0; k < 9; ++k) {
      float v = bp[k];
      bfv[k] = (__bf16)((v > 0.f) ? 1.f : ((v < 0.f) ? -1.f : 0.f));
    }

    acc = __builtin_amdgcn_wmma_f32_16x16x32_bf16(
        false, a, false, bfv, (short)0, acc, false, false);
    __syncthreads();
  }

  // ---- writeback raw conv out: VGPR r -> M = r (+8 for hi half), N = lane&15
#pragma unroll
  for (int r = 0; r < 8; ++r) {
    int b = mt * 16 + r + hi * 8;
    int o = brow;
    float v = acc[r] * scaleW + bias[o * (HH * WW) + hw];
    out[(b * COUT + o) * (HH * WW) + hw] = v;
  }
}

// ---------------------------------------------------------------- batchnorm
__global__ __launch_bounds__(256) void k_bnstats(const float* __restrict__ out,
                                                 const float* __restrict__ gamma,
                                                 const float* __restrict__ beta,
                                                 float* __restrict__ ws) {
  __shared__ float r1[256], r2[256];
  const int o = blockIdx.x;
  float s = 0.f, sq = 0.f;
  for (int i = threadIdx.x; i < BB * HH * WW; i += 256) {
    int b = i >> 10, hw = i & 1023;
    float v = out[((b * COUT + o) << 10) + hw];
    s += v;
    sq += v * v;
  }
  r1[threadIdx.x] = s;
  r2[threadIdx.x] = sq;
  __syncthreads();
  for (int st = 128; st > 0; st >>= 1) {
    if ((int)threadIdx.x < st) {
      r1[threadIdx.x] += r1[threadIdx.x + st];
      r2[threadIdx.x] += r2[threadIdx.x + st];
    }
    __syncthreads();
  }
  if (threadIdx.x == 0) {
    const float n = (float)(BB * HH * WW);
    float mean = r1[0] / n;
    float var = r2[0] / n - mean * mean;      // biased variance
    float rstd = rsqrtf(var + BN_EPS);
    float g = gamma[o] * rstd;
    ws[WS_CHA + o] = g;
    ws[WS_CHB + o] = beta[o] - g * mean;
  }
}

__global__ __launch_bounds__(256) void k_bnrelu(float* __restrict__ out,
                                                float* __restrict__ ws) {
  __shared__ float red[256];
  float s = 0.f;
  for (long i = (long)blockIdx.x * 256 + threadIdx.x; i < NTOT;
       i += (long)NB_RED * 256) {
    int o = ((int)i >> 10) & 63;
    float y = ws[WS_CHA + o] * out[i] + ws[WS_CHB + o];
    y = fmaxf(y, 0.f);
    out[i] = y;
    s += y;                                   // |relu(y)| == relu(y)
  }
  red[threadIdx.x] = s;
  __syncthreads();
  for (int st = 128; st > 0; st >>= 1) {
    if ((int)threadIdx.x < st) red[threadIdx.x] += red[threadIdx.x + st];
    __syncthreads();
  }
  if (threadIdx.x == 0) ws[WS_P5 + blockIdx.x] = red[0];
}

__global__ __launch_bounds__(256) void k_binout(float* __restrict__ out,
                                                const float* __restrict__ ws) {
  const float sc = ws[WS_SCALEOUT];
  for (long i = (long)blockIdx.x * 256 + threadIdx.x; i < NTOT;
       i += (long)NB_RED * 256)
    out[i] = (out[i] > 0.f) ? sc : 0.f;       // sign(relu(y)) * scale
}

// ---------------------------------------------------------------- launch
extern "C" void kernel_launch(void* const* d_in, const int* in_sizes, int n_in,
                              void* d_out, int out_size, void* d_ws,
                              size_t ws_size, hipStream_t stream) {
  const float* x     = (const float*)d_in[0];
  const float* w     = (const float*)d_in[1];
  const float* biasp = (const float*)d_in[2];
  const float* gamma = (const float*)d_in[3];
  const float* beta  = (const float*)d_in[4];
  float* out = (float*)d_out;
  float* ws  = (float*)d_ws;

  // 1) scale_w = mean(|w|)
  k_absw<<<NB_RED, 256, 0, stream>>>(w, ws);
  k_finalize<<<1, 256, 0, stream>>>(ws + WS_P1, ws + WS_SCALEW,
                                    1.0f / (float)NWEL);
  // 2) locally-connected binary conv via WMMA + TDM weight pipeline
  k_conv_wmma<<<HH * WW, 256, LDS_TOTAL, stream>>>(x, w, biasp, ws, out);
  // 3) BN batch stats per channel
  k_bnstats<<<COUT, 256, 0, stream>>>(out, gamma, beta, ws);
  // 4) BN + ReLU in place, accumulate mean(|y|)
  k_bnrelu<<<NB_RED, 256, 0, stream>>>(out, ws);
  k_finalize<<<1, 256, 0, stream>>>(ws + WS_P5, ws + WS_SCALEOUT,
                                    1.0f / (float)NTOT);
  // 5) output binarization
  k_binout<<<NB_RED, 256, 0, stream>>>(out, ws);
}
